// HeadwiseLowRankModule_12378095747711
// MI455X (gfx1250) — compile-verified, matrix operand order checked by reading, not run on hardware
//
#include <hip/hip_runtime.h>
#include <stdint.h>

// ---------------- problem dims ----------------
#define M_TOT   8192      // B * L
#define N1      2048      // NUM_GROUPS * RANK
#define K1      4096      // D_IN
#define NGROUP  8
#define RANK_   256
#define GDIM    512
#define D_OUT   4096
#define THREADS 256

// ---------------- fast-path tiling (async bf16 operands, 3-stage pipeline) ----------------
#define F_BM  128
#define F_BN  256
#define F_BK  64
#define F_BKP 72          // padded LDS stride: 36 dwords -> conflict-free frag reads
#define F_ASZ (F_BM * F_BKP)              // elements per A stage buffer
#define F_BSZ (F_BN * F_BKP)              // elements per B stage buffer
#define F_STAGES 3
#define F_SMEM_BYTES ((F_STAGES * (F_ASZ + F_BSZ)) * 2)   // 165888 B of the 320 KB WGP LDS
#define F_ASYNC_PER_STAGE 12              // 4 (A) + 8 (B) async instrs per wave per stage

// ---------------- fallback tiling (fused f32->bf16 convert) ----------------
#define S_BM  128
#define S_BN  128
#define S_BK  32
#define S_BKP 40

typedef __attribute__((ext_vector_type(16))) __bf16 bf16x16;
typedef __attribute__((ext_vector_type(8)))  __bf16 bf16x8;
typedef __attribute__((ext_vector_type(4)))  __bf16 bf16x4;
typedef __attribute__((ext_vector_type(8)))  float  f32x8;

// Low 32 bits of a generic shared-aperture address == LDS byte offset (ISA 10.2)
static __device__ __forceinline__ unsigned lds_addr32(const void* p) {
  return (unsigned)(uintptr_t)p;
}

// D = A x B + C ; A operand carries the WMMA-M dim (rows across lanes 0-15),
// B operand carries the WMMA-N dim. We feed the WEIGHT tile as A and the
// ACTIVATION tile as B so each lane ends up holding 8 consecutive output
// columns -> vectorized b128 epilogue stores.
static __device__ __forceinline__ f32x8 wmma_bf16(bf16x16 a, bf16x16 b, f32x8 c) {
  return __builtin_amdgcn_wmma_f32_16x16x32_bf16(false, a, false, b, (short)0, c, false, false);
}

// WMMA 16-bit operand fragment (ISA 7.12.2); valid for both operands since
// both tiles are row-major with K contiguous.
template <int LDK>
static __device__ __forceinline__ bf16x16 lds_frag(const __bf16* T, int row0, int k, int lane) {
  int rr = row0 + (lane & 15);
  int kk = k + ((lane & 16) >> 1);           // +8 for upper half-wave
  const __bf16* p = T + rr * LDK + kk;
  bf16x8 lo = *(const bf16x8*)(p);           // K = kk .. kk+7
  bf16x8 hi = *(const bf16x8*)(p + 16);      // K = kk+16 .. kk+23
  return __builtin_shufflevector(lo, hi, 0,1,2,3,4,5,6,7,8,9,10,11,12,13,14,15);
}

// =========================================================================
// FAST PATH
// =========================================================================

// copy ROWS x 64 bf16 tile (128 B rows) global -> LDS, padded stride F_BKP
template <int ROWS>
static __device__ __forceinline__ void async_tile(unsigned lds_base,
                                                  const __bf16* __restrict__ gbase,
                                                  int ld, int tid) {
#pragma unroll
  for (int i = 0; i < (ROWS * 8) / THREADS; ++i) {
    int flat = tid + i * THREADS;
    int row = flat >> 3;                 // 8 x b128 per row
    int c   = (flat & 7) << 3;           // element offset in row
    unsigned dst = lds_base + (unsigned)((row * F_BKP + c) * 2);
    unsigned off = (unsigned)((row * ld + c) * 2);   // byte offset (GVS addressing)
    asm volatile("global_load_async_to_lds_b128 %0, %1, %2"
                 :: "v"(dst), "v"(off), "s"(gbase) : "memory");
  }
}

static __device__ __forceinline__ void issue_stage(unsigned aBase, unsigned bBase, int buf,
                                                   const __bf16* Ap, int lda,
                                                   const __bf16* Bp, int ldb,
                                                   int s, int tid) {
  async_tile<F_BM>(aBase + (unsigned)(buf * F_ASZ * 2), Ap + (long)s * F_BK, lda, tid);
  async_tile<F_BN>(bBase + (unsigned)(buf * F_BSZ * 2), Bp + (long)s * F_BK, ldb, tid);
}

// one 32-deep K step: 4x4 grid of 16x16 wmma per wave (64(m) x 64(n) wave tile)
// Am = activation tile (m rows) -> WMMA B operand; Wn = weight tile (n rows) -> WMMA A operand
static __device__ __forceinline__ void mma_step64(const __bf16* Am, const __bf16* Wn,
                                                  f32x8 (&acc)[4][4], int wm, int wn,
                                                  int k, int lane) {
  bf16x16 an[4], bm[4];
#pragma unroll
  for (int i = 0; i < 4; ++i) an[i] = lds_frag<F_BKP>(Wn, wn + i * 16, k, lane);
#pragma unroll
  for (int j = 0; j < 4; ++j) bm[j] = lds_frag<F_BKP>(Am, wm + j * 16, k, lane);
#pragma unroll
  for (int ni = 0; ni < 4; ++ni)
#pragma unroll
    for (int mi = 0; mi < 4; ++mi)
      acc[ni][mi] = wmma_bf16(an[ni], bm[mi], acc[ni][mi]);
}

__global__ __launch_bounds__(THREADS, 1) void cvt_f32_to_bf16(
    const float* __restrict__ src, __bf16* __restrict__ dst, int n4) {
  int i = blockIdx.x * THREADS + threadIdx.x;
  if (i >= n4) return;
  float4 v = ((const float4*)src)[i];
  bf16x4 o;
  o[0] = (__bf16)v.x; o[1] = (__bf16)v.y; o[2] = (__bf16)v.z; o[3] = (__bf16)v.w;
  ((bf16x4*)dst)[i] = o;
}

// latents[8192,2048](bf16) = Xb[8192,4096] @ Vb[2048,4096]^T
__global__ __launch_bounds__(THREADS, 1) void gemm1_async(
    const __bf16* __restrict__ Xb, const __bf16* __restrict__ Vb,
    __bf16* __restrict__ Lat) {
  extern __shared__ __align__(16) __bf16 smem[];   // [A0|A1|A2|B0|B1|B2]
  const int tid = threadIdx.x, lane = tid & 31, wave = tid >> 5;
  const int wm = (wave & 1) << 6;        // 2 waves along M (64 rows)
  const int wn = (wave >> 1) << 6;       // 4 waves along N (64 cols)
  const long m0 = (long)blockIdx.y * F_BM;
  const long n0 = (long)blockIdx.x * F_BN;
  const __bf16* Ap = Xb + m0 * K1;
  const __bf16* Bp = Vb + n0 * K1;

  f32x8 acc[4][4];
#pragma unroll
  for (int ni = 0; ni < 4; ++ni)
#pragma unroll
    for (int mi = 0; mi < 4; ++mi)
#pragma unroll
      for (int e = 0; e < 8; ++e) acc[ni][mi][e] = 0.0f;

  const unsigned aBase = lds_addr32(smem);
  const unsigned bBase = aBase + (unsigned)(F_STAGES * F_ASZ * 2);

  const int S = K1 / F_BK;   // 64 stages
  issue_stage(aBase, bBase, 0, Ap, K1, Bp, K1, 0, tid);
  issue_stage(aBase, bBase, 1, Ap, K1, Bp, K1, 1, tid);

#pragma unroll 1
  for (int s = 0; s < S; ++s) {
    // retire stage s (async completes in order; newest stage leaves <=12 pending)
    if (s + 1 < S) asm volatile("s_wait_asynccnt 12" ::: "memory");
    else           asm volatile("s_wait_asynccnt 0"  ::: "memory");
    __syncthreads();   // all waves: stage-s data ready AND stage-(s-1) reads done
    if (s + 2 < S)
      issue_stage(aBase, bBase, (s + 2) % F_STAGES, Ap, K1, Bp, K1, s + 2, tid);
    const __bf16* Ac = smem + (s % F_STAGES) * F_ASZ;
    const __bf16* Wc = smem + F_STAGES * F_ASZ + (s % F_STAGES) * F_BSZ;
    mma_step64(Ac, Wc, acc, wm, wn, 0, lane);
    mma_step64(Ac, Wc, acc, wm, wn, 32, lane);
  }

  // D layout after swap: lane holds m = lane&15; elems = 8 consecutive n (hi half +8)
#pragma unroll
  for (int ni = 0; ni < 4; ++ni) {
    long nb = n0 + wn + ni * 16 + ((lane & 16) >> 1);
#pragma unroll
    for (int mi = 0; mi < 4; ++mi) {
      long m = m0 + wm + mi * 16 + (lane & 15);
      bf16x8 v;
#pragma unroll
      for (int e = 0; e < 8; ++e) v[e] = (__bf16)acc[ni][mi][e];
      *(bf16x8*)(Lat + m * N1 + nb) = v;
    }
  }
}

// out[8192,4096](f32) = latents_g @ Ub16[g]^T + U_b[g], head-permuted stores
__global__ __launch_bounds__(THREADS, 1) void gemm2_async(
    const __bf16* __restrict__ Lat, const __bf16* __restrict__ Ub16,
    const float* __restrict__ Ub, float* __restrict__ Out) {
  extern __shared__ __align__(16) __bf16 smem[];
  const int tid = threadIdx.x, lane = tid & 31, wave = tid >> 5;
  const int wm = (wave & 1) << 6, wn = (wave >> 1) << 6;
  const int g  = blockIdx.z;
  const long m0 = (long)blockIdx.y * F_BM;
  const int n0 = blockIdx.x * F_BN;                        // {0, 256} within group
  const __bf16* Ap = Lat + m0 * N1 + (long)g * RANK_;      // ld = N1
  const __bf16* Bp = Ub16 + ((long)g * GDIM + n0) * RANK_; // ld = RANK_

  f32x8 acc[4][4];
#pragma unroll
  for (int ni = 0; ni < 4; ++ni)
#pragma unroll
    for (int mi = 0; mi < 4; ++mi)
#pragma unroll
      for (int e = 0; e < 8; ++e) acc[ni][mi][e] = 0.0f;

  const unsigned aBase = lds_addr32(smem);
  const unsigned bBase = aBase + (unsigned)(F_STAGES * F_ASZ * 2);

  const int S = RANK_ / F_BK;  // 4 stages
  issue_stage(aBase, bBase, 0, Ap, N1, Bp, RANK_, 0, tid);
  issue_stage(aBase, bBase, 1, Ap, N1, Bp, RANK_, 1, tid);

#pragma unroll 1
  for (int s = 0; s < S; ++s) {
    if (s + 1 < S) asm volatile("s_wait_asynccnt 12" ::: "memory");
    else           asm volatile("s_wait_asynccnt 0"  ::: "memory");
    __syncthreads();
    if (s + 2 < S)
      issue_stage(aBase, bBase, (s + 2) % F_STAGES, Ap, N1, Bp, RANK_, s + 2, tid);
    const __bf16* Ac = smem + (s % F_STAGES) * F_ASZ;
    const __bf16* Wc = smem + F_STAGES * F_ASZ + (s % F_STAGES) * F_BSZ;
    mma_step64(Ac, Wc, acc, wm, wn, 0, lane);
    mma_step64(Ac, Wc, acc, wm, wn, 32, lane);
  }

  // bias + static head permutation p -> h = 8*(p%4) + p/4 ; 8 consecutive n per lane
#pragma unroll
  for (int ni = 0; ni < 4; ++ni) {
    int nlb = n0 + wn + ni * 16 + ((lane & 16) >> 1);   // base of 8 consecutive n, [0,512)
    int p   = (g << 2) + (nlb >> 7);
    int h   = ((p & 3) << 3) + (p >> 2);
    int colb = (h << 7) + (nlb & 127);
    float4 b0 = *(const float4*)(Ub + g * GDIM + nlb);
    float4 b1 = *(const float4*)(Ub + g * GDIM + nlb + 4);
#pragma unroll
    for (int mi = 0; mi < 4; ++mi) {
      long m = m0 + wm + mi * 16 + (lane & 15);
      float4 lo, hi;
      lo.x = acc[ni][mi][0] + b0.x; lo.y = acc[ni][mi][1] + b0.y;
      lo.z = acc[ni][mi][2] + b0.z; lo.w = acc[ni][mi][3] + b0.w;
      hi.x = acc[ni][mi][4] + b1.x; hi.y = acc[ni][mi][5] + b1.y;
      hi.z = acc[ni][mi][6] + b1.z; hi.w = acc[ni][mi][7] + b1.w;
      *(float4*)(Out + m * (long)D_OUT + colb)     = lo;
      *(float4*)(Out + m * (long)D_OUT + colb + 4) = hi;
    }
  }
}

// =========================================================================
// FALLBACK PATH (small workspace): fused f32->bf16 convert, register staging
// =========================================================================

static __device__ __forceinline__ void g_load_tile_f32(const float* __restrict__ src,
                                                       int ld, float4 (&r)[4], int tid) {
#pragma unroll
  for (int i = 0; i < 4; ++i) {
    int flat = tid + i * THREADS;
    int row = flat >> 3;
    int c4  = flat & 7;
    r[i] = *(const float4*)(src + (long)row * ld + (c4 << 2));
  }
}

static __device__ __forceinline__ void lds_store_tile(__bf16* dst, const float4 (&r)[4], int tid) {
#pragma unroll
  for (int i = 0; i < 4; ++i) {
    int flat = tid + i * THREADS;
    int row = flat >> 3;
    int c4  = flat & 7;
    bf16x4 v;
    v[0] = (__bf16)r[i].x; v[1] = (__bf16)r[i].y;
    v[2] = (__bf16)r[i].z; v[3] = (__bf16)r[i].w;
    *(bf16x4*)(dst + row * S_BKP + (c4 << 2)) = v;
  }
}

// wave tile 32(m) x 64(n); weight rows feed WMMA-A, activation rows feed WMMA-B
static __device__ __forceinline__ void mma_tile32(const __bf16* Am, const __bf16* Wn,
                                                  f32x8 (&acc)[4][2], int wm, int wn, int lane) {
  bf16x16 an[4];
#pragma unroll
  for (int j = 0; j < 4; ++j) an[j] = lds_frag<S_BKP>(Wn, wn + j * 16, 0, lane);
  bf16x16 bm0 = lds_frag<S_BKP>(Am, wm,      0, lane);
  bf16x16 bm1 = lds_frag<S_BKP>(Am, wm + 16, 0, lane);
#pragma unroll
  for (int ni = 0; ni < 4; ++ni) {
    acc[ni][0] = wmma_bf16(an[ni], bm0, acc[ni][0]);
    acc[ni][1] = wmma_bf16(an[ni], bm1, acc[ni][1]);
  }
}

__global__ __launch_bounds__(THREADS) void gemm1_fb(
    const float* __restrict__ X, const float* __restrict__ VT, __bf16* __restrict__ Lat) {
  __shared__ __align__(16) __bf16 As[2][S_BM * S_BKP];
  __shared__ __align__(16) __bf16 Bs[2][S_BN * S_BKP];
  const int tid = threadIdx.x, lane = tid & 31, wave = tid >> 5;
  const int wm = (wave >> 1) << 5, wn = (wave & 1) << 6;
  const long m0 = (long)blockIdx.y * S_BM;
  const long n0 = (long)blockIdx.x * S_BN;
  const float* Xp = X + m0 * K1;
  const float* Vp = VT + n0 * K1;

  f32x8 acc[4][2];
#pragma unroll
  for (int ni = 0; ni < 4; ++ni)
#pragma unroll
    for (int mi = 0; mi < 2; ++mi)
#pragma unroll
      for (int e = 0; e < 8; ++e) acc[ni][mi][e] = 0.0f;

  float4 ra[4], rb[4];
  g_load_tile_f32(Xp, K1, ra, tid);
  g_load_tile_f32(Vp, K1, rb, tid);
  lds_store_tile(As[0], ra, tid);
  lds_store_tile(Bs[0], rb, tid);
  __syncthreads();

  const int S = K1 / S_BK;
#pragma unroll 1
  for (int s = 0; s < S; ++s) {
    const int cur = s & 1;
    const bool pre = (s + 1) < S;
    if (pre) {
      g_load_tile_f32(Xp + (s + 1) * S_BK, K1, ra, tid);
      g_load_tile_f32(Vp + (s + 1) * S_BK, K1, rb, tid);
    }
    mma_tile32(As[cur], Bs[cur], acc, wm, wn, lane);
    if (pre) {
      lds_store_tile(As[cur ^ 1], ra, tid);
      lds_store_tile(Bs[cur ^ 1], rb, tid);
    }
    __syncthreads();
  }

#pragma unroll
  for (int ni = 0; ni < 4; ++ni) {
    long nb = n0 + wn + ni * 16 + ((lane & 16) >> 1);
#pragma unroll
    for (int mi = 0; mi < 2; ++mi) {
      long m = m0 + wm + mi * 16 + (lane & 15);
      bf16x8 v;
#pragma unroll
      for (int e = 0; e < 8; ++e) v[e] = (__bf16)acc[ni][mi][e];
      *(bf16x8*)(Lat + m * N1 + nb) = v;
    }
  }
}

__global__ __launch_bounds__(THREADS) void gemm2_fb(
    const __bf16* __restrict__ Lat, const float* __restrict__ U,
    const float* __restrict__ Ub, float* __restrict__ Out) {
  __shared__ __align__(16) __bf16 As[2][S_BM * S_BKP];
  __shared__ __align__(16) __bf16 Bs[2][S_BN * S_BKP];
  const int tid = threadIdx.x, lane = tid & 31, wave = tid >> 5;
  const int wm = (wave >> 1) << 5, wn = (wave & 1) << 6;
  const int g  = blockIdx.z;
  const long m0 = (long)blockIdx.y * S_BM;
  const int n0 = blockIdx.x * S_BN;
  const __bf16* Ap = Lat + m0 * N1 + (long)g * RANK_;
  const float*  Up = U + ((long)g * GDIM + n0) * RANK_;

  f32x8 acc[4][2];
#pragma unroll
  for (int ni = 0; ni < 4; ++ni)
#pragma unroll
    for (int mi = 0; mi < 2; ++mi)
#pragma unroll
      for (int e = 0; e < 8; ++e) acc[ni][mi][e] = 0.0f;

  float4 rb[4];
#pragma unroll
  for (int i = 0; i < 2; ++i) {
    int flat = tid + i * THREADS;
    int row = flat >> 2;
    int c   = (flat & 3) << 3;
    unsigned dst = lds_addr32(&As[0][0]) + (unsigned)((row * S_BKP + c) * 2);
    unsigned off = (unsigned)(((long)row * N1 + c) * 2);
    asm volatile("global_load_async_to_lds_b128 %0, %1, %2"
                 :: "v"(dst), "v"(off), "s"(Ap) : "memory");
  }
  g_load_tile_f32(Up, RANK_, rb, tid);
  lds_store_tile(Bs[0], rb, tid);
  asm volatile("s_wait_asynccnt 0" ::: "memory");
  __syncthreads();

  const int S = RANK_ / S_BK;
#pragma unroll 1
  for (int s = 0; s < S; ++s) {
    const int cur = s & 1;
    const bool pre = (s + 1) < S;
    if (pre) {
      const __bf16* An = Ap + (s + 1) * S_BK;
#pragma unroll
      for (int i = 0; i < 2; ++i) {
        int flat = tid + i * THREADS;
        int row = flat >> 2;
        int c   = (flat & 3) << 3;
        unsigned dst = lds_addr32(&As[cur ^ 1][0]) + (unsigned)((row * S_BKP + c) * 2);
        unsigned off = (unsigned)(((long)row * N1 + c) * 2);
        asm volatile("global_load_async_to_lds_b128 %0, %1, %2"
                     :: "v"(dst), "v"(off), "s"(An) : "memory");
      }
      g_load_tile_f32(Up + (s + 1) * S_BK, RANK_, rb, tid);
    }
    mma_tile32(As[cur], Bs[cur], acc, wm, wn, lane);
    if (pre) {
      lds_store_tile(Bs[cur ^ 1], rb, tid);
      asm volatile("s_wait_asynccnt 0" ::: "memory");
    }
    __syncthreads();
  }

#pragma unroll
  for (int ni = 0; ni < 4; ++ni) {
    int nlb = n0 + wn + ni * 16 + ((lane & 16) >> 1);
    int p   = (g << 2) + (nlb >> 7);
    int h   = ((p & 3) << 3) + (p >> 2);
    int colb = (h << 7) + (nlb & 127);
    float4 b0 = *(const float4*)(Ub + g * GDIM + nlb);
    float4 b1 = *(const float4*)(Ub + g * GDIM + nlb + 4);
#pragma unroll
    for (int mi = 0; mi < 2; ++mi) {
      long m = m0 + wm + mi * 16 + (lane & 15);
      float4 lo, hi;
      lo.x = acc[ni][mi][0] + b0.x; lo.y = acc[ni][mi][1] + b0.y;
      lo.z = acc[ni][mi][2] + b0.z; lo.w = acc[ni][mi][3] + b0.w;
      hi.x = acc[ni][mi][4] + b1.x; hi.y = acc[ni][mi][5] + b1.y;
      hi.z = acc[ni][mi][6] + b1.z; hi.w = acc[ni][mi][7] + b1.w;
      *(float4*)(Out + m * (long)D_OUT + colb)     = lo;
      *(float4*)(Out + m * (long)D_OUT + colb + 4) = hi;
    }
  }
}

// =========================================================================
extern "C" void kernel_launch(void* const* d_in, const int* in_sizes, int n_in,
                              void* d_out, int out_size, void* d_ws, size_t ws_size,
                              hipStream_t stream) {
  (void)in_sizes; (void)n_in; (void)out_size;
  const float* x    = (const float*)d_in[0];
  const float* VT_w = (const float*)d_in[1];
  const float* U_w  = (const float*)d_in[2];
  const float* U_b  = (const float*)d_in[3];
  float* out = (float*)d_out;

  const size_t xN = (size_t)M_TOT * K1;            // 33,554,432
  const size_t vN = (size_t)N1 * K1;               //  8,388,608
  const size_t uN = (size_t)NGROUP * GDIM * RANK_; //  1,048,576
  const size_t lN = (size_t)M_TOT * N1;            // 16,777,216
  const size_t need_fast = (xN + vN + uN + lN) * sizeof(__bf16);   // ~120 MiB

  if (ws_size >= need_fast) {
    __bf16* xb  = (__bf16*)d_ws;
    __bf16* vb  = xb + xN;
    __bf16* ub  = vb + vN;
    __bf16* lat = ub + uN;

    cvt_f32_to_bf16<<<dim3((unsigned)(xN / 4 / THREADS)), THREADS, 0, stream>>>(x, xb, (int)(xN / 4));
    cvt_f32_to_bf16<<<dim3((unsigned)(vN / 4 / THREADS)), THREADS, 0, stream>>>(VT_w, vb, (int)(vN / 4));
    cvt_f32_to_bf16<<<dim3((unsigned)(uN / 4 / THREADS)), THREADS, 0, stream>>>(U_w, ub, (int)(uN / 4));

    dim3 g1(N1 / F_BN, M_TOT / F_BM);            // 8 x 64
    gemm1_async<<<g1, THREADS, F_SMEM_BYTES, stream>>>(xb, vb, lat);
    dim3 g2(GDIM / F_BN, M_TOT / F_BM, NGROUP);  // 2 x 64 x 8
    gemm2_async<<<g2, THREADS, F_SMEM_BYTES, stream>>>(lat, ub, U_b, out);
  } else {
    __bf16* lat = (__bf16*)d_ws;                 // 32 MiB
    dim3 g1(N1 / S_BN, M_TOT / S_BM);            // 16 x 64
    gemm1_fb<<<g1, THREADS, 0, stream>>>(x, VT_w, lat);
    dim3 g2(GDIM / S_BN, M_TOT / S_BM, NGROUP);  // 4 x 64 x 8
    gemm2_fb<<<g2, THREADS, 0, stream>>>(lat, U_w, U_b, out);
  }
}
